// MHSelfAttention_77309411786
// MI455X (gfx1250) — matrix operand order repaired and checked by reading, not compiled, not run on hardware
//
#include <hip/hip_runtime.h>
#include <hip/hip_bf16.h>

typedef __attribute__((ext_vector_type(16))) _Float16 v16h;
typedef __attribute__((ext_vector_type(8)))  _Float16 v8h;
typedef __attribute__((ext_vector_type(8)))  float    v8f;
typedef __attribute__((ext_vector_type(4)))  float    v4f;
typedef __attribute__((ext_vector_type(4)))  int      v4i;

#define BB 2
#define TT 4096
#define EE 512
#define HH 8
#define SS 64

// ---------- fragment loaders (contiguous k order k = 16*hi + s, same for A and B) ----------
// A (16x32 f16): lane<16: row=l, K = k0+0..15 (v[0..15])
//                lane>=16: row=l, K = k0+16..31
static __device__ inline v16h load_a_f16(const _Float16* base, int row0, int k0,
                                         int ld, int l, int hi) {
  const _Float16* p = base + (size_t)(row0 + l) * ld + k0 + hi * 16;
  v8h x0 = *(const v8h*)(p);
  v8h x1 = *(const v8h*)(p + 8);
  v16h a;
#pragma unroll
  for (int i = 0; i < 8; ++i) { a[i] = x0[i]; a[8 + i] = x1[i]; }
  return a;
}

// B (32x16 f16): lane = column; 16 contiguous K at p (k-offset hi*16 applied by caller)
static __device__ inline v16h ld_b16(const _Float16* p) {
  v8h x0 = *(const v8h*)(p);
  v8h x1 = *(const v8h*)(p + 8);
  v16h b;
#pragma unroll
  for (int i = 0; i < 8; ++i) { b[i] = x0[i]; b[8 + i] = x1[i]; }
  return b;
}

static __device__ inline v16h load_a_f32(const float* base, int row0, int k0,
                                         int ld, int l, int hi) {
  const float* p = base + (size_t)(row0 + l) * ld + k0 + hi * 16;
  v4f x0 = *(const v4f*)(p);
  v4f x1 = *(const v4f*)(p + 4);
  v4f x2 = *(const v4f*)(p + 8);
  v4f x3 = *(const v4f*)(p + 12);
  v16h a;
#pragma unroll
  for (int i = 0; i < 4; ++i) {
    a[i]      = (_Float16)x0[i];
    a[4 + i]  = (_Float16)x1[i];
    a[8 + i]  = (_Float16)x2[i];
    a[12 + i] = (_Float16)x3[i];
  }
  return a;
}

static __device__ inline v16h load_b_f32(const float* base, int n0, int k0,
                                         int ld, int l, int hi) {
  const float* p = base + (size_t)(n0 + l) * ld + k0 + hi * 16;
  v16h b;
#pragma unroll
  for (int i = 0; i < 16; i += 4) {
    v4f x = *(const v4f*)(p + i);
    b[i] = (_Float16)x[0]; b[i + 1] = (_Float16)x[1];
    b[i + 2] = (_Float16)x[2]; b[i + 3] = (_Float16)x[3];
  }
  return b;
}

// ---------- 16B global -> LDS copy: async (ASYNCcnt) if available ----------
static __device__ inline void cp16_g2l(const _Float16* g, _Float16* l) {
#if __has_builtin(__builtin_amdgcn_global_load_async_to_lds_b128)
  v4i* gp = (v4i*)g;  // const-cast + reinterpret (generic)
  v4i* lp = (v4i*)l;
  __builtin_amdgcn_global_load_async_to_lds_b128(
      (__attribute__((address_space(1))) v4i*)gp,
      (__attribute__((address_space(3))) v4i*)lp, 0, 0);
#else
  *(v8h*)l = *(const v8h*)g;
#endif
}

static __device__ inline void wait_async0() {
#if __has_builtin(__builtin_amdgcn_global_load_async_to_lds_b128)
#if __has_builtin(__builtin_amdgcn_s_wait_asynccnt)
  __builtin_amdgcn_s_wait_asynccnt(0);
#else
  asm volatile("s_wait_asynccnt 0" ::: "memory");
#endif
#endif
}

// ---------- kernel 1: QKV projection ----------
// grid (BT/16=512, 8, 3), block 128 (4 waves); wave computes one 16x16 tile.
__global__ __launch_bounds__(128) void proj_qkv(
    const float* __restrict__ x,
    const float* __restrict__ Wq, const float* __restrict__ Wk,
    const float* __restrict__ Wv,
    _Float16* __restrict__ Qh, _Float16* __restrict__ Kh,
    _Float16* __restrict__ Vt) {
  const int wave = threadIdx.x >> 5, lane = threadIdx.x & 31;
  const int l = lane & 15, hi = lane >> 4;
  const int row0 = blockIdx.x * 16;                 // token tile (over B*T)
  const int n0 = (blockIdx.y * 4 + wave) * 16;      // output-feature tile
  const int mat = blockIdx.z;
  const float* W = (mat == 0) ? Wq : ((mat == 1) ? Wk : Wv);
  v8f c = {};
#pragma unroll 4
  for (int k0 = 0; k0 < EE; k0 += 32) {
    v16h a = load_a_f32(x, row0, k0, EE, l, hi);
    v16h bf = load_b_f32(W, n0, k0, EE, l, hi);
    c = __builtin_amdgcn_wmma_f32_16x16x32_f16(false, a, false, bf,
                                               (short)0, c, false, false);
  }
  const float qscale = 0.04419417382415922f;  // 512^-0.5 (both e^-0.25 folded into Q)
#pragma unroll
  for (int r = 0; r < 8; ++r) {
    int row = row0 + r + 8 * hi;          // global token index b*T + t
    int col = n0 + l;                     // output feature
    int bb = row >> 12, t = row & (TT - 1);
    int h = col >> 6, s = col & (SS - 1);
    float v = c[r];
    size_t idx = (((size_t)bb * HH + h) * TT + t) * SS + s;     // (b,h,t,s)
    if (mat == 0)      Qh[idx] = (_Float16)(v * qscale);
    else if (mat == 1) Kh[idx] = (_Float16)v;
    else               Vt[(((size_t)bb * HH + h) * SS + s) * TT + t] = (_Float16)v; // (b,h,s,t)
  }
}

// ---------- kernel 2: flash attention, LDS-staged K/V, double buffered ----------
// grid (T/64=64, B*H=16), block 128 (4 waves); wave owns a 16x64 Q strip.
__global__ __launch_bounds__(128) void attn_flash(
    const _Float16* __restrict__ Qh, const _Float16* __restrict__ Kh,
    const _Float16* __restrict__ Vt, const int* __restrict__ pmask,
    _Float16* __restrict__ Oh) {
  __shared__ __align__(16) _Float16 Ksh[2][64][72];  // [buf][key][s]
  __shared__ __align__(16) _Float16 Vsh[2][64][72];  // [buf][s][key]
  __shared__ __align__(16) _Float16 psh[4][16][72];  // per-wave P staging tile
  const int wave = threadIdx.x >> 5, lane = threadIdx.x & 31;
  const int l = lane & 15, hi = lane >> 4;
  const int qb = blockIdx.x;
  const int bh = blockIdx.y;
  const int b = bh >> 3, h = bh & 7;
  const _Float16* Qb = Qh + (size_t)bh * TT * SS;
  const _Float16* Kb = Kh + (size_t)bh * TT * SS;
  const _Float16* Vb = Vt + (size_t)bh * SS * TT;
  const int qrow0 = qb * 64 + wave * 16;

  // stage a 64-key tile (K rows + V columns) into LDS buffer `buf`
  auto issue_tile = [&](int jblk, int buf) {
    const int kbn = jblk * 64;
#pragma unroll
    for (int i = 0; i < 4; ++i) {
      int c = (int)threadIdx.x + i * 128;      // 512 chunks of 16B per matrix
      int row = c >> 3, off = (c & 7) * 8;
      cp16_g2l(Kb + (size_t)(kbn + row) * SS + off, &Ksh[buf][row][off]);
      cp16_g2l(Vb + (size_t)row * TT + kbn + off, &Vsh[buf][row][off]);
    }
  };

  v16h aq0 = load_a_f16(Qb, qrow0, 0, SS, l, hi);
  v16h aq1 = load_a_f16(Qb, qrow0, 32, SS, l, hi);

  float rbias[8], rmax[8], rsum[8];
#pragma unroll
  for (int r = 0; r < 8; ++r) {
    int qq = qrow0 + r + 8 * hi;
    rbias[r] = (pmask[b * TT + qq] != 0) ? 0.f : -1e30f;  // query-dim mask (as reference)
    rmax[r] = -1e30f;
    rsum[r] = 0.f;
  }
  v8f oacc[4] = {v8f{}, v8f{}, v8f{}, v8f{}};

  issue_tile(0, 0);
  for (int j = 0; j <= qb; ++j) {
    const int cur = j & 1;
    wait_async0();          // my copies for buffer `cur` are done
    __syncthreads();        // everyone's copies done; prev iteration reads done
    if (j < qb) issue_tile(j + 1, cur ^ 1);

    // ---- scores S = Q * K^T  (16x64 per wave) ----
    v8f sc[4];
#pragma unroll
    for (int nt = 0; nt < 4; ++nt) {
      v8f c = {};
      c = __builtin_amdgcn_wmma_f32_16x16x32_f16(
          false, aq0, false, ld_b16(&Ksh[cur][nt * 16 + l][hi * 16]),
          (short)0, c, false, false);
      c = __builtin_amdgcn_wmma_f32_16x16x32_f16(
          false, aq1, false, ld_b16(&Ksh[cur][nt * 16 + l][32 + hi * 16]),
          (short)0, c, false, false);
      sc[nt] = c;
    }
    // ---- causal + padding masks ----
    const bool diag = (j == qb);
#pragma unroll
    for (int nt = 0; nt < 4; ++nt)
#pragma unroll
      for (int r = 0; r < 8; ++r) {
        float v = sc[nt][r] + rbias[r];
        if (diag && (nt * 16 + l) > (wave * 16 + r + 8 * hi)) v = -1e30f;
        sc[nt][r] = v;
      }
    // ---- online softmax: row stats live in the row's 16-lane half ----
    float nmax[8], corr[8], lsum[8];
#pragma unroll
    for (int r = 0; r < 8; ++r) {
      float m = fmaxf(fmaxf(sc[0][r], sc[1][r]), fmaxf(sc[2][r], sc[3][r]));
#pragma unroll
      for (int off = 1; off < 16; off <<= 1) m = fmaxf(m, __shfl_xor(m, off, 32));
      nmax[r] = fmaxf(rmax[r], m);
      corr[r] = __expf(rmax[r] - nmax[r]);
      rmax[r] = nmax[r];
      lsum[r] = 0.f;
    }
#pragma unroll
    for (int nt = 0; nt < 4; ++nt)
#pragma unroll
      for (int r = 0; r < 8; ++r) {
        float p = __expf(sc[nt][r] - nmax[r]);
        sc[nt][r] = p;
        lsum[r] += p;
      }
#pragma unroll
    for (int r = 0; r < 8; ++r) {
#pragma unroll
      for (int off = 1; off < 16; off <<= 1) lsum[r] += __shfl_xor(lsum[r], off, 32);
      rsum[r] = rsum[r] * corr[r] + lsum[r];
    }
#pragma unroll
    for (int nt = 0; nt < 4; ++nt)
#pragma unroll
      for (int r = 0; r < 8; ++r) oacc[nt][r] *= corr[r];

    // ---- C-layout -> A-layout for P via per-wave LDS tile ----
#pragma unroll
    for (int nt = 0; nt < 4; ++nt)
#pragma unroll
      for (int r = 0; r < 8; ++r)
        psh[wave][r + 8 * hi][nt * 16 + l] = (_Float16)sc[nt][r];
    asm volatile("s_wait_dscnt 0" ::: "memory");  // same-wave LDS RAW
    v16h pa0, pa1;
    {
      v8h x0 = *(const v8h*)&psh[wave][l][hi * 16];
      v8h x1 = *(const v8h*)&psh[wave][l][hi * 16 + 8];
      v8h y0 = *(const v8h*)&psh[wave][l][32 + hi * 16];
      v8h y1 = *(const v8h*)&psh[wave][l][32 + hi * 16 + 8];
#pragma unroll
      for (int i = 0; i < 8; ++i) {
        pa0[i] = x0[i]; pa0[8 + i] = x1[i];
        pa1[i] = y0[i]; pa1[8 + i] = y1[i];
      }
    }
    // ---- O += P * V  (Vsh[s][key]: contiguous keys per s-column) ----
#pragma unroll
    for (int nt = 0; nt < 4; ++nt) {
      oacc[nt] = __builtin_amdgcn_wmma_f32_16x16x32_f16(
          false, pa0, false, ld_b16(&Vsh[cur][nt * 16 + l][hi * 16]),
          (short)0, oacc[nt], false, false);
      oacc[nt] = __builtin_amdgcn_wmma_f32_16x16x32_f16(
          false, pa1, false, ld_b16(&Vsh[cur][nt * 16 + l][32 + hi * 16]),
          (short)0, oacc[nt], false, false);
    }
  }
  // ---- normalize and store O as (b, t, h*64+s) f16 ----
#pragma unroll
  for (int nt = 0; nt < 4; ++nt)
#pragma unroll
    for (int r = 0; r < 8; ++r) {
      int row = qrow0 + r + 8 * hi;
      int col = h * 64 + nt * 16 + l;
      Oh[((size_t)b * TT + row) * EE + col] = (_Float16)(oacc[nt][r] / rsum[r]);
    }
}

// ---------- kernel 3: output projection ----------
__global__ __launch_bounds__(128) void proj_out(
    const _Float16* __restrict__ Oh, const float* __restrict__ Wu,
    const float* __restrict__ bu, float* __restrict__ out) {
  const int wave = threadIdx.x >> 5, lane = threadIdx.x & 31;
  const int l = lane & 15, hi = lane >> 4;
  const int row0 = blockIdx.x * 16;
  const int n0 = (blockIdx.y * 4 + wave) * 16;
  v8f c = {};
#pragma unroll 4
  for (int k0 = 0; k0 < EE; k0 += 32) {
    v16h a = load_a_f16(Oh, row0, k0, EE, l, hi);
    v16h bf = load_b_f32(Wu, n0, k0, EE, l, hi);
    c = __builtin_amdgcn_wmma_f32_16x16x32_f16(false, a, false, bf,
                                               (short)0, c, false, false);
  }
  const float bias = bu[n0 + l];
#pragma unroll
  for (int r = 0; r < 8; ++r) {
    int row = row0 + r + 8 * hi;
    out[(size_t)row * EE + n0 + l] = c[r] + bias;
  }
}

extern "C" void kernel_launch(void* const* d_in, const int* in_sizes, int n_in,
                              void* d_out, int out_size, void* d_ws,
                              size_t ws_size, hipStream_t stream) {
  (void)in_sizes; (void)n_in; (void)out_size; (void)ws_size;
  const float* x  = (const float*)d_in[0];
  const int*   pm = (const int*)d_in[1];
  const float* Wq = (const float*)d_in[2];
  const float* Wk = (const float*)d_in[3];
  const float* Wv = (const float*)d_in[4];
  const float* Wu = (const float*)d_in[5];
  const float* bu = (const float*)d_in[6];
  float* out = (float*)d_out;

  const size_t nqkv = (size_t)BB * HH * TT * SS;  // 4 Mi halfs each
  _Float16* Qh = (_Float16*)d_ws;
  _Float16* Kh = Qh + nqkv;
  _Float16* Vt = Kh + nqkv;
  _Float16* Oh = Vt + nqkv;                       // total ~33.6 MB of ws

  proj_qkv<<<dim3((BB * TT) / 16, EE / 64, 3), 128, 0, stream>>>(
      x, Wq, Wk, Wv, Qh, Kh, Vt);
  attn_flash<<<dim3(TT / 64, BB * HH), 128, 0, stream>>>(Qh, Kh, Vt, pm, Oh);
  proj_out<<<dim3((BB * TT) / 16, EE / 64), 128, 0, stream>>>(Oh, Wu, bu, out);
}
